// MultiHeadAttention_72370198938219
// MI455X (gfx1250) — matrix operand order repaired and checked by reading, not compiled, not run on hardware
//
#include <hip/hip_runtime.h>
#include <hip/hip_bf16.h>

typedef __attribute__((ext_vector_type(16))) _Float16 v16h;
typedef __attribute__((ext_vector_type(8)))  _Float16 v8h;
typedef __attribute__((ext_vector_type(8)))  float    v8f;

#define KB_  8
#define KS_  2048
#define KD_  1024
#define KH_  16
#define KDH_ 64

#define BM 128
#define BN 64
#define BK 32

union V16U { v16h v; v8h h[2]; };

struct TileRegs { float4 a[4]; float4 b[2]; };

// ---------------------------------------------------------------------------
// Load one (BMxBK, BKxBN) tile pair from global fp32 into registers.
// `full` is block-uniform -> scalar branch, no exec-mask predication on the
// fast path (all dominant GEMMs here are tile-aligned).
// ---------------------------------------------------------------------------
__device__ __forceinline__ void load_tile(
    TileRegs& t, const float* __restrict__ Ap, const float* __restrict__ Bp,
    int lda, int ldb, int bm, int bn, int k0, int M, int N,
    int arow, int acol, int bk, int bn0, bool full)
{
    const float4* asrc = (const float4*)(Ap + (long)(bm + arow) * lda + k0 + acol);
    const float4* bsrc = (const float4*)(Bp + (long)(k0 + bk) * ldb + bn + bn0);
    if (full) {
        #pragma unroll
        for (int j = 0; j < 4; ++j) t.a[j] = asrc[j];
        #pragma unroll
        for (int j = 0; j < 2; ++j) t.b[j] = bsrc[j];
    } else {
        const bool ok = (bm + arow) < M;
        #pragma unroll
        for (int j = 0; j < 4; ++j)
            t.a[j] = ok ? asrc[j] : make_float4(0.f, 0.f, 0.f, 0.f);
        #pragma unroll
        for (int j = 0; j < 2; ++j)
            t.b[j] = (bn + bn0 + 4 * j) < N ? bsrc[j] : make_float4(0.f, 0.f, 0.f, 0.f);
    }
}

// fp32 regs -> f16 LDS (A row-major [BM][BK], B transposed [BN][BK+8])
__device__ __forceinline__ void store_tile(
    _Float16 (*__restrict__ sA)[BK], _Float16 (*__restrict__ sB)[BK + 8],
    const TileRegs& t, int arow, int acol, int bk, int bn0)
{
    #pragma unroll
    for (int j = 0; j < 4; ++j) {
        sA[arow][acol + 4 * j + 0] = (_Float16)t.a[j].x;
        sA[arow][acol + 4 * j + 1] = (_Float16)t.a[j].y;
        sA[arow][acol + 4 * j + 2] = (_Float16)t.a[j].z;
        sA[arow][acol + 4 * j + 3] = (_Float16)t.a[j].w;
    }
    #pragma unroll
    for (int j = 0; j < 2; ++j) {
        sB[bn0 + 4 * j + 0][bk] = (_Float16)t.b[j].x;
        sB[bn0 + 4 * j + 1][bk] = (_Float16)t.b[j].y;
        sB[bn0 + 4 * j + 2][bk] = (_Float16)t.b[j].z;
        sB[bn0 + 4 * j + 3][bk] = (_Float16)t.b[j].w;
    }
}

// ---------------------------------------------------------------------------
// Generic batched mixed-precision GEMM: C = A @ B, fp32 in/out, f16 WMMA core,
// double-buffered LDS (one barrier per K-step; global loads for tile k+1 are
// issued before the WMMAs of tile k so HBM latency hides behind matrix math).
// Per-batch offsets are two-level: off = (bz/Hdiv)*s?b + (bz%Hdiv)*s?h.
// Block tile 128x64, 8 wave32 waves (4x2), each wave 32x32 via 2x2 WMMA.
// K must be a multiple of 32 (true here: 1024/32/64/2048).
// ---------------------------------------------------------------------------
__global__ __launch_bounds__(256) void gemm_wmma_f16(
    const float* __restrict__ Abase, long sAb, long sAh, int lda,
    const float* __restrict__ Bbase, long sBb, long sBh, int ldb,
    float* __restrict__ Cbase, long sCb, long sCh, int ldc,
    int M, int N, int K, int Hdiv)
{
    __shared__ __align__(16) _Float16 sA[2][BM][BK];      // 2 x 8 KB
    __shared__ __align__(16) _Float16 sB[2][BN][BK + 8];  // 2 x 5 KB

    const int bz = blockIdx.z;
    const int bb = bz / Hdiv;
    const int bh = bz - bb * Hdiv;
    const float* Ap = Abase + (long)bb * sAb + (long)bh * sAh;
    const float* Bp = Bbase + (long)bb * sBb + (long)bh * sBh;
    float*       Cp = Cbase + (long)bb * sCb + (long)bh * sCh;

    const int bm   = blockIdx.x * BM;
    const int bn   = blockIdx.y * BN;
    const int tid  = threadIdx.x;
    const int wave = tid >> 5;
    const int lane = tid & 31;
    const int wm   = (wave & 3) * 32;
    const int wn   = (wave >> 2) * 32;
    const int lm   = lane & 15;
    const int hi   = lane >> 4;

    const int arow = tid >> 1;        // 0..127
    const int acol = (tid & 1) * 16;  // 0 or 16
    const int bk   = tid >> 3;        // 0..31
    const int bn0  = (tid & 7) * 8;   // 0..56

    const bool full = (bm + BM <= M) && (bn + BN <= N);

    v8f acc[2][2] = {};
    TileRegs t;

    // prologue: stage tile 0 into buffer 0
    load_tile(t, Ap, Bp, lda, ldb, bm, bn, 0, M, N, arow, acol, bk, bn0, full);
    store_tile(sA[0], sB[0], t, arow, acol, bk, bn0);
    __syncthreads();

    int p = 0;
    for (int k0 = 0; k0 < K; k0 += BK) {
        const bool more = (k0 + BK) < K;
        // issue global loads for the NEXT tile first (latency overlap)
        if (more)
            load_tile(t, Ap, Bp, lda, ldb, bm, bn, k0 + BK, M, N,
                      arow, acol, bk, bn0, full);

        // ---- fragments from current buffer, per ISA 16-bit layouts ----
        // A (16x32 MxK): lane m=lm; elems 0-7: K=8*hi..+7, elems 8-15: 16+8*hi..+7
        V16U af[2], bf[2];
        #pragma unroll
        for (int tm = 0; tm < 2; ++tm) {
            const _Float16* base = &sA[p][wm + tm * 16 + lm][0];
            af[tm].h[0] = *(const v8h*)(base + 8 * hi);
            af[tm].h[1] = *(const v8h*)(base + 16 + 8 * hi);
        }
        // B (32x16 KxN): lane n=lm; elems 0-7: K=8*hi..+7, elems 8-15: 16+8*hi..+7
        #pragma unroll
        for (int tn = 0; tn < 2; ++tn) {
            const _Float16* base = &sB[p][wn + tn * 16 + lm][0];
            bf[tn].h[0] = *(const v8h*)(base + 8 * hi);
            bf[tn].h[1] = *(const v8h*)(base + 16 + 8 * hi);
        }
        #pragma unroll
        for (int tm = 0; tm < 2; ++tm)
            #pragma unroll
            for (int tn = 0; tn < 2; ++tn)
                acc[tm][tn] = __builtin_amdgcn_wmma_f32_16x16x32_f16(
                    false, af[tm].v, false, bf[tn].v,
                    (short)0, acc[tm][tn], false, false);

        // stage next tile into the other buffer (no conflict with readers of p)
        if (more)
            store_tile(sA[p ^ 1], sB[p ^ 1], t, arow, acol, bk, bn0);
        __syncthreads();
        p ^= 1;
    }

    // ---- epilogue: C/D layout — VGPR r: lanes 0-15 M=r, lanes 16-31 M=8+r ----
    if (full) {
        #pragma unroll
        for (int tm = 0; tm < 2; ++tm)
            #pragma unroll
            for (int tn = 0; tn < 2; ++tn) {
                const int col = bn + wn + tn * 16 + lm;
                #pragma unroll
                for (int r = 0; r < 8; ++r) {
                    const int row = bm + wm + tm * 16 + hi * 8 + r;
                    Cp[(long)row * ldc + col] = acc[tm][tn][r];
                }
            }
    } else {
        #pragma unroll
        for (int tm = 0; tm < 2; ++tm)
            #pragma unroll
            for (int tn = 0; tn < 2; ++tn) {
                const int col = bn + wn + tn * 16 + lm;
                if (col < N) {
                    #pragma unroll
                    for (int r = 0; r < 8; ++r) {
                        const int row = bm + wm + tm * 16 + hi * 8 + r;
                        if (row < M) Cp[(long)row * ldc + col] = acc[tm][tn][r];
                    }
                }
            }
    }
}

// ---------------------------------------------------------------------------
// 64x64 transpose of A (for stage-2 Q @ A^T as an NN GEMM)
// ---------------------------------------------------------------------------
__global__ void transpose64(const float* __restrict__ A, float* __restrict__ AT)
{
    int t = threadIdx.x + blockIdx.x * blockDim.x;
    if (t < 64 * 64) {
        int f = t >> 6, d = t & 63;
        AT[(long)d * 64 + f] = A[t];  // AT[d][f] = A[f][d]
    }
}

// ---------------------------------------------------------------------------
// Stage-1 softmax over S (column softmax of SC=[bh][S][64]), transposed write
// into attn=[bh][64][S] (the [B,H,DH,S] output, reused as GEMM operand).
// Block = 256 threads organized (64 d) x (4 s-strides); coalesced 64-wide reads.
// ---------------------------------------------------------------------------
__global__ __launch_bounds__(256) void softmax_stage1(
    const float* __restrict__ SC, float* __restrict__ attn, float scale)
{
    const int bh = blockIdx.x;
    const int tx = threadIdx.x & 63;  // d
    const int ty = threadIdx.x >> 6;  // 0..3
    const float* sc = SC + (long)bh * KS_ * KDH_;
    float* at = attn + (long)bh * KDH_ * KS_;
    __shared__ float red[4][64];

    float mx = -3.402823466e38f;
    for (int s = ty; s < KS_; s += 4)
        mx = fmaxf(mx, sc[(long)s * KDH_ + tx] * scale);
    red[ty][tx] = mx;
    __syncthreads();
    mx = fmaxf(fmaxf(red[0][tx], red[1][tx]), fmaxf(red[2][tx], red[3][tx]));
    __syncthreads();

    float sum = 0.f;
    for (int s = ty; s < KS_; s += 4)
        sum += __expf(sc[(long)s * KDH_ + tx] * scale - mx);
    red[ty][tx] = sum;
    __syncthreads();
    sum = red[0][tx] + red[1][tx] + red[2][tx] + red[3][tx];
    const float inv = 1.0f / sum;

    for (int s = ty; s < KS_; s += 4)
        at[(long)tx * KS_ + s] = __expf(sc[(long)s * KDH_ + tx] * scale - mx) * inv;
}

// ---------------------------------------------------------------------------
// Stage-2 softmax: in-place row softmax, rows of 64 floats, one wave32 per row
// (lane handles d=lane and d=lane+32; shfl_xor reductions).
// ---------------------------------------------------------------------------
__global__ __launch_bounds__(256) void softmax_rows64(
    float* __restrict__ X, long nrows, float scale)
{
    long row = (long)blockIdx.x * 8 + (threadIdx.x >> 5);
    const int lane = threadIdx.x & 31;
    if (row >= nrows) return;
    float* x = X + row * 64;
    float v0 = x[lane] * scale;
    float v1 = x[lane + 32] * scale;
    float mx = fmaxf(v0, v1);
    #pragma unroll
    for (int off = 16; off > 0; off >>= 1) mx = fmaxf(mx, __shfl_xor(mx, off, 32));
    float e0 = __expf(v0 - mx), e1 = __expf(v1 - mx);
    float s = e0 + e1;
    #pragma unroll
    for (int off = 16; off > 0; off >>= 1) s += __shfl_xor(s, off, 32);
    const float inv = 1.0f / s;
    x[lane]      = e0 * inv;
    x[lane + 32] = e1 * inv;
}

// ---------------------------------------------------------------------------
// out = LayerNorm(X + R) over rows of D=1024; one 256-thread block per row.
// ---------------------------------------------------------------------------
__global__ __launch_bounds__(256) void add_layernorm(
    const float* __restrict__ X, const float* __restrict__ R,
    const float* __restrict__ g, const float* __restrict__ b,
    float* __restrict__ out)
{
    const long row = blockIdx.x;
    const int tid = threadIdx.x;
    const float* x = X + row * KD_;
    const float* r = R + row * KD_;
    float* o = out + row * KD_;
    __shared__ float s1[256], s2[256];

    float vals[4], lsum = 0.f, lsq = 0.f;
    #pragma unroll
    for (int j = 0; j < 4; ++j) {
        const int c = tid + j * 256;
        const float v = x[c] + r[c];
        vals[j] = v;
        lsum += v;
        lsq += v * v;
    }
    s1[tid] = lsum; s2[tid] = lsq;
    __syncthreads();
    for (int off = 128; off > 0; off >>= 1) {
        if (tid < off) { s1[tid] += s1[tid + off]; s2[tid] += s2[tid + off]; }
        __syncthreads();
    }
    const float mean = s1[0] * (1.0f / KD_);
    const float var  = s2[0] * (1.0f / KD_) - mean * mean;
    const float rstd = rsqrtf(var + 1e-5f);
    #pragma unroll
    for (int j = 0; j < 4; ++j) {
        const int c = tid + j * 256;
        o[c] = (vals[j] - mean) * rstd * g[c] + b[c];
    }
}

// ---------------------------------------------------------------------------
extern "C" void kernel_launch(void* const* d_in, const int* in_sizes, int n_in,
                              void* d_out, int out_size, void* d_ws, size_t ws_size,
                              hipStream_t stream)
{
    (void)in_sizes; (void)n_in; (void)out_size; (void)ws_size;

    const float* Xq   = (const float*)d_in[0];
    const float* Xk   = (const float*)d_in[1];
    const float* Xv   = (const float*)d_in[2];
    const float* Wq1  = (const float*)d_in[3];
    const float* Wq2  = (const float*)d_in[4];
    const float* Wk1  = (const float*)d_in[5];
    const float* Wk2  = (const float*)d_in[6];
    const float* Wv   = (const float*)d_in[7];
    const float* Wfc  = (const float*)d_in[8];
    const float* Amat = (const float*)d_in[9];
    const float* ln_g = (const float*)d_in[10];
    const float* ln_b = (const float*)d_in[11];

    const long MS = (long)KB_ * KS_;   // 16384 total rows
    const long BH = (long)KB_ * KH_;   // 128 (batch*heads)

    // workspace layout (floats); big regions reused once their producer dies
    float* ws = (float*)d_ws;
    long off = 0;
    float* AT    = ws + off; off += 4096;
    float* Qmid  = ws + off; off += MS * 32;
    float* Kmid  = ws + off; off += MS * 32;
    float* Qproj = ws + off; off += MS * KD_;
    float* Kproj = ws + off; off += MS * KD_;
    float* Vproj = ws + off; off += MS * KD_;
    float* SC    = ws + off; off += BH * KS_ * KDH_;   // scores scratch (both stages)
    float* VV    = ws + off; off += BH * KDH_ * KDH_;
    float* VVV  = Kproj;   // Kproj dead after stage-1 scores GEMM
    float* out1 = Qproj;   // Qproj dead after stage-2 scores GEMM
    float* out2 = Vproj;   // Vproj dead after VV GEMM

    float* outF  = (float*)d_out;                    // outputs [B,S,D]
    float* attn1 = outF + (long)KB_ * KS_ * KD_;     // attn_VV [B,H,DH,S]

    const float scale = 0.125f;  // 1/sqrt(64)

    auto gemm = [&](const float* A, long sAb, long sAh, int lda,
                    const float* B, long sBb, long sBh, int ldb,
                    float* C, long sCb, long sCh, int ldc,
                    int M, int N, int K, int batch, int Hdiv) {
        dim3 grid((M + BM - 1) / BM, (N + BN - 1) / BN, batch);
        gemm_wmma_f16<<<grid, dim3(256), 0, stream>>>(
            A, sAb, sAh, lda, B, sBb, sBh, ldb, C, sCb, sCh, ldc, M, N, K, Hdiv);
    };

    transpose64<<<16, 256, 0, stream>>>(Amat, AT);

    // projections
    gemm(Xq,   0, 0, KD_, Wq1, 0, 0, 32,  Qmid,  0, 0, 32,  (int)MS, 32,  KD_, 1, 1);
    gemm(Xk,   0, 0, KD_, Wk1, 0, 0, 32,  Kmid,  0, 0, 32,  (int)MS, 32,  KD_, 1, 1);
    gemm(Qmid, 0, 0, 32,  Wq2, 0, 0, KD_, Qproj, 0, 0, KD_, (int)MS, KD_, 32,  1, 1);
    gemm(Kmid, 0, 0, 32,  Wk2, 0, 0, KD_, Kproj, 0, 0, KD_, (int)MS, KD_, 32,  1, 1);
    gemm(Xv,   0, 0, KD_, Wv,  0, 0, KD_, Vproj, 0, 0, KD_, (int)MS, KD_, KD_, 1, 1);

    // stage 1: SC[bh][s][d] = Kproj_bh @ A  (== scores_VV transposed)
    gemm(Kproj, (long)KS_ * KD_, KDH_, KD_,
         Amat,  0, 0, KDH_,
         SC,    (long)KH_ * KS_ * KDH_, (long)KS_ * KDH_, KDH_,
         KS_, KDH_, KDH_, (int)BH, KH_);
    softmax_stage1<<<(int)BH, 256, 0, stream>>>(SC, attn1, scale);

    // VV[bh] = attn_bh (64 x S) @ Vproj_bh (S x 64)
    gemm(attn1, (long)KH_ * KDH_ * KS_, (long)KDH_ * KS_, KS_,
         Vproj, (long)KS_ * KD_, KDH_, KD_,
         VV,    (long)KH_ * KDH_ * KDH_, (long)KDH_ * KDH_, KDH_,
         KDH_, KDH_, KS_, (int)BH, KH_);

    // stage 2: SC[bh][s][d] = Qproj_bh @ A^T
    gemm(Qproj, (long)KS_ * KD_, KDH_, KD_,
         AT,    0, 0, KDH_,
         SC,    (long)KH_ * KS_ * KDH_, (long)KS_ * KDH_, KDH_,
         KS_, KDH_, KDH_, (int)BH, KH_);
    softmax_rows64<<<(int)(BH * KS_ / 8), 256, 0, stream>>>(SC, BH * KS_, scale);

    // VVV[bh] = attn2_bh (S x 64) @ VV_bh (64 x 64)  -> contiguous [B,H,S,DH]
    gemm(SC, (long)KH_ * KS_ * KDH_, (long)KS_ * KDH_, KDH_,
         VV, (long)KH_ * KDH_ * KDH_, (long)KDH_ * KDH_, KDH_,
         VVV, (long)KH_ * KS_ * KDH_, (long)KS_ * KDH_, KDH_,
         KS_, KDH_, KDH_, (int)BH, KH_);

    // faithful reshape == linear view of [B,H,S,DH] per batch; then LN(res+x)
    add_layernorm<<<(int)MS, 256, 0, stream>>>(VVV, Xq, ln_g, ln_b, out1);
    gemm(out1, 0, 0, KD_, Wfc, 0, 0, KD_, out2, 0, 0, KD_, (int)MS, KD_, KD_, 1, 1);
    add_layernorm<<<(int)MS, 256, 0, stream>>>(out2, out1, ln_g, ln_b, outF);
}
